// SSNet_44641890074811
// MI455X (gfx1250) — compile-verified, matrix operand order checked by reading, not run on hardware
//
#include <hip/hip_runtime.h>
#include <hip/hip_bf16.h>
#include <math.h>

// ---------------------------------------------------------------------------
// Types for CDNA5 WMMA
// ---------------------------------------------------------------------------
typedef __attribute__((ext_vector_type(16))) __bf16 v16bf;
typedef __attribute__((ext_vector_type(8)))  float  v8f;
typedef __attribute__((ext_vector_type(2)))  float  v2f;

union ABfrag { v16bf v; unsigned u[8]; };

__device__ __forceinline__ unsigned f2bf(float f) {
  // round-to-nearest-even f32 -> bf16 (bits in low 16)
  unsigned u = __float_as_uint(f);
  u += 0x7FFFu + ((u >> 16) & 1u);
  return u >> 16;
}

static inline int cdiv(int a, int b) { return (a + b - 1) / b; }

// ---------------------------------------------------------------------------
// Zero fill (for atomic-accumulated fc1 output)
// ---------------------------------------------------------------------------
__global__ void k_zero(float* __restrict__ p, int n) {
  int i = blockIdx.x * blockDim.x + threadIdx.x;
  if (i < n) p[i] = 0.0f;
}

// ---------------------------------------------------------------------------
// Direct 5x5 conv, Cin=1, BN folded, ReLU.  out: [B][Cout][H-4][W-4]
// ---------------------------------------------------------------------------
__global__ void k_conv5_c1(const float* __restrict__ x, const float* __restrict__ w,
                           const float* __restrict__ bi, const float* __restrict__ g,
                           const float* __restrict__ be, const float* __restrict__ mn,
                           const float* __restrict__ vr,
                           float* __restrict__ out, int B, int Cout, int H, int W) {
  int OH = H - 4, OW = W - 4;
  int total = B * Cout * OH * OW;
  int idx = blockIdx.x * blockDim.x + threadIdx.x;
  if (idx >= total) return;
  int ox = idx % OW; int t = idx / OW;
  int oy = t % OH;   t /= OH;
  int c  = t % Cout; int b = t / Cout;

  float scale = g[c] * rsqrtf(vr[c] + 1e-5f);
  float bias  = bi[c] * scale + be[c] - mn[c] * scale;

  const float* xp = x + ((size_t)b * H + oy) * W + ox;
  const float* wp = w + c * 25;
  float sum = 0.0f;
#pragma unroll
  for (int r = 0; r < 5; ++r)
#pragma unroll
    for (int cc = 0; cc < 5; ++cc)
      sum += xp[r * W + cc] * wp[r * 5 + cc];
  float val = sum * scale + bias;
  out[(((size_t)b * Cout + c) * OH + oy) * OW + ox] = fmaxf(val, 0.0f);
}

// ---------------------------------------------------------------------------
// Generic max pool (VALID), optional per-channel bias + ReLU fused after max.
// in:[B][C][IH][IW] -> out:[B][Ctot][..][..] at channel offset c0
// ---------------------------------------------------------------------------
__global__ void k_pool(const float* __restrict__ in, float* __restrict__ out,
                       int B, int C, int IH, int IW, int k, int s, int OH, int OW,
                       int Ctot, int c0, const float* __restrict__ bias, int do_relu) {
  int total = B * C * OH * OW;
  int idx = blockIdx.x * blockDim.x + threadIdx.x;
  if (idx >= total) return;
  int ox = idx % OW; int t = idx / OW;
  int oy = t % OH;   t /= OH;
  int c  = t % C;    int b = t / C;
  const float* p = in + (((size_t)b * C + c) * IH + oy * s) * IW + ox * s;
  float mx = -INFINITY;
  for (int dy = 0; dy < k; ++dy)
    for (int dx = 0; dx < k; ++dx)
      mx = fmaxf(mx, p[dy * IW + dx]);
  if (bias) mx += bias[c];
  if (do_relu) mx = fmaxf(mx, 0.0f);
  out[(((size_t)b * Ctot + c0 + c) * OH + oy) * OW + ox] = mx;
}

// ---------------------------------------------------------------------------
// ROI adaptive max pool.  roiX:[2][3][252][252], ROI:[2][5][1][4] int32.
// Writes h channels 20..34 of [2][35][125][125]; batch order reversed.
// ---------------------------------------------------------------------------
__global__ void k_roipool(const float* __restrict__ roiX, const int* __restrict__ ROI,
                          float* __restrict__ h) {
  const int total = 2 * 15 * 125 * 125;
  int idx = blockIdx.x * blockDim.x + threadIdx.x;
  if (idx >= total) return;
  int ox = idx % 125; int t = idx / 125;
  int oy = t % 125;   t /= 125;
  int c  = t % 15;    int bo = t / 15;
  int k = 1 - bo;          // crops[1] + crops[0]
  int j = c / 5, i = c % 5;
  const int* rp = ROI + (k * 5 + i) * 4;
  // int(v*0.9) for v>=0 == (9v)/10
  int y1 = (rp[0] * 9) / 10, y2 = (rp[1] * 9) / 10;
  int x1 = (rp[2] * 9) / 10, x2 = (rp[3] * 9) / 10;
  int Hc = y2 - y1, Wc = x2 - x1;
  int rs = (oy * Hc) / 125, re = ((oy + 1) * Hc + 124) / 125;
  int cs = (ox * Wc) / 125, ce = ((ox + 1) * Wc + 124) / 125;
  const float* p = roiX + (((size_t)k * 3 + j) * 252 + y1) * 252 + x1;
  float mx = -INFINITY;
  for (int r = rs; r < re; ++r)
    for (int cc = cs; cc < ce; ++cc)
      mx = fmaxf(mx, p[r * 252 + cc]);
  h[(((size_t)bo * 35 + 20 + c) * 125 + oy) * 125 + ox] = mx;
}

// ---------------------------------------------------------------------------
// Fold BN scale into conv weights, pack bf16 as K-pair dwords: Wp[k/2][Npad].
// Also emits folded bias. w layout [Cout][Cin][5][5] => k = ci*25 + r*5 + cc.
// ---------------------------------------------------------------------------
__global__ void k_prep_w(const float* __restrict__ w, const float* __restrict__ bi,
                         const float* __restrict__ g, const float* __restrict__ be,
                         const float* __restrict__ mn, const float* __restrict__ vr,
                         unsigned* __restrict__ Wp, float* __restrict__ biasf,
                         int Cout, int Kreal, int Kpad, int Npad) {
  int total = (Kpad / 2) * Npad;
  int idx = blockIdx.x * blockDim.x + threadIdx.x;
  if (idx >= total) return;
  int n = idx % Npad, kp = idx / Npad;
  unsigned lo = 0, hi = 0;
  if (n < Cout) {
    float scale = g[n] * rsqrtf(vr[n] + 1e-5f);
    int k = 2 * kp;
    if (k     < Kreal) lo = f2bf(w[(size_t)n * Kreal + k    ] * scale);
    if (k + 1 < Kreal) hi = f2bf(w[(size_t)n * Kreal + k + 1] * scale);
    if (kp == 0) biasf[n] = bi[n] * scale + be[n] - mn[n] * scale;
  } else if (kp == 0) {
    biasf[n] = 0.0f;
  }
  Wp[idx] = lo | (hi << 16);
}

// ---------------------------------------------------------------------------
// Implicit-GEMM 5x5 VALID conv via v_wmma_f32_16x16x32_bf16.
// in:[B][Cin][H][W], Wp packed [Kpad/2][Npad] (bf16 K-pairs), out:[B][Cout][OH][OW]
// Workgroup covers MT*16 M-rows (output pixels) x Npad columns.
// Wave w owns N-tile w and iterates MT M-tiles, reusing one B fragment per
// K-step across MT WMMAs (4x fewer B loads, 4x WMMA density per step).
// blockDim.x must be (Npad/16)*32.
// ---------------------------------------------------------------------------
template <int MT>
__global__ void k_conv_wmma(const float* __restrict__ in, const unsigned* __restrict__ Wp,
                            float* __restrict__ out,
                            int Cin, int H, int W, int OH, int OW,
                            int Kreal, int Kpad, int Npad, int Cout) {
  __shared__ unsigned sA[MT * 16 * 16];   // MT x (16 rows x 16 K-pair dwords)
  const int b    = blockIdx.y;
  const int m0   = blockIdx.x * (16 * MT);
  const int wave = threadIdx.x >> 5;
  const int lane = threadIdx.x & 31;
  const int M    = OH * OW;
  const float* inB = in + (size_t)b * Cin * H * W;

  v8f acc[MT];
#pragma unroll
  for (int tt = 0; tt < MT; ++tt) acc[tt] = (v8f){};

  const int half = lane >> 4;
  const int mi   = lane & 15;
  const int n    = wave * 16 + (lane & 15);

  for (int k0 = 0; k0 < Kpad; k0 += 32) {
    __syncthreads();
    // Stage im2col A-tile (bf16 pairs) into LDS: rows 0..MT*16-1
    for (int t = (int)threadIdx.x; t < MT * 256; t += (int)blockDim.x) {
      int row = t >> 4;        // M row within super-tile
      int kp  = t & 15;        // K-pair index
      int m = m0 + row;
      unsigned lo = 0, hi = 0;
      if (m < M) {
        int oy = m / OW, ox = m % OW;
        int k = k0 + 2 * kp;
        if (k < Kreal) {
          int ci = k / 25, rc = k % 25;
          lo = f2bf(inB[((size_t)ci * H + oy + rc / 5) * W + ox + rc % 5]);
        }
        if (k + 1 < Kreal) {
          int k1 = k + 1;
          int ci = k1 / 25, rc = k1 % 25;
          hi = f2bf(inB[((size_t)ci * H + oy + rc / 5) * W + ox + rc % 5]);
        }
      }
      sA[t] = lo | (hi << 16);
    }
    __syncthreads();

    // B fragment once per K-step (documented 16-bit 16x32 layout:
    // lanes 0-15: v0..3 -> K 0..7, v4..7 -> K 16..23; lanes 16-31: K 8..15, 24..31)
    ABfrag bb;
#pragma unroll
    for (int j = 0; j < 8; ++j) {
      int kidx = (j < 4) ? (half * 8 + 2 * j) : (16 + half * 8 + 2 * (j - 4));
      bb.u[j] = Wp[((size_t)((k0 + kidx) >> 1)) * Npad + n];
    }
    // MT WMMAs sharing the B fragment
#pragma unroll
    for (int tt = 0; tt < MT; ++tt) {
      ABfrag a;
#pragma unroll
      for (int j = 0; j < 8; ++j) {
        int kidx = (j < 4) ? (half * 8 + 2 * j) : (16 + half * 8 + 2 * (j - 4));
        a.u[j] = sA[(tt * 16 + mi) * 16 + (kidx >> 1)];
      }
      acc[tt] = __builtin_amdgcn_wmma_f32_16x16x32_bf16(
          false, a.v, false, bb.v, (short)0, acc[tt], false, false);
    }
  }

  // C/D layout: VGPR r -> M = r + 8*half, N = lane&15
  if (n < Cout) {
#pragma unroll
    for (int tt = 0; tt < MT; ++tt) {
#pragma unroll
      for (int r = 0; r < 8; ++r) {
        int m = m0 + tt * 16 + r + 8 * half;
        if (m < M) out[((size_t)b * Cout + n) * M + m] = acc[tt][r];
      }
    }
  }
}

// ---------------------------------------------------------------------------
// fc1: [2,19600] @ [19600,1024] via v_wmma_f32_16x16x4_f32, K split across
// blockIdx.y with f32 atomic accumulation into pre-zeroed h1:[2][1024].
// One wave per block; M padded 2->16. HBM-bound on fw1 (80 MB).
//
// Key point: A rows >= 2 only influence D rows >= 2, which are discarded, so
// lanes mi>=2 simply re-read row 0 (valid memory) — no masking, no exec
// divergence. Main loop is fully guard-free pointer streaming; a <=2-element
// guarded tail handles the chunk remainder. Uniform-condition prefetch pulls
// the weight stream ~128KB ahead (global_prefetch_b8).
// ---------------------------------------------------------------------------
__global__ void k_fc1_wmma(const float* __restrict__ h4, const float* __restrict__ fw1,
                           float* __restrict__ h1) {
  const int lane  = threadIdx.x & 31;
  const int ntile = blockIdx.x;      // 0..63
  const int kc    = blockIdx.y;      // 0..7
  const int K = 19600;
  const int kper = (K + 7) / 8;      // 2450
  const int kbeg = kc * kper;
  int kend = kbeg + kper; if (kend > K) kend = K;

  const int half = lane >> 4;
  const int mi   = lane & 15;
  const int n    = ntile * 16 + mi;

  // 32-bit A 16x4 layout: lanes 0-15 hold K=0,1 in v0,v1; lanes 16-31 hold K=2,3.
  // Row pointer: mi==1 -> batch row 1; all other lanes read batch row 0
  // (rows >=2 of the padded A are don't-care for the retained D rows 0,1).
  const float* ap = h4 + ((mi == 1) ? (size_t)K : 0) + kbeg + 2 * half;
  const float* bp = fw1 + (size_t)(kbeg + 2 * half) * 1024 + n;

  v8f acc = {};
  const int steps = (kend - kbeg) >> 2;     // full 4-wide K steps (612)
  for (int s = 0; s < steps; ++s) {
    float a0 = ap[0];
    float a1 = ap[1];
    float b0 = bp[0];
    float b1 = bp[1024];
    if (s + 8 < steps)   // uniform cond -> scalar branch; ~128KB ahead
      __builtin_prefetch(bp + 32 * 1024, 0, 1);
    v2f av = {a0, a1};
    v2f bv = {b0, b1};
    acc = __builtin_amdgcn_wmma_f32_16x16x4_f32(
        false, av, false, bv, (short)0, acc, false, false);
    ap += 4;
    bp += 4 * 1024;
  }
  // tail (kper % 4 == 2): at most one guarded step
  for (int k0 = kbeg + steps * 4; k0 < kend; k0 += 4) {
    int ka = k0 + 2 * half;
    float a0 = (ka     < kend) ? ap[0] : 0.0f;
    float a1 = (ka + 1 < kend) ? ap[1] : 0.0f;
    float b0 = (ka     < kend) ? bp[0]    : 0.0f;
    float b1 = (ka + 1 < kend) ? bp[1024] : 0.0f;
    v2f av = {a0, a1};
    v2f bv = {b0, b1};
    acc = __builtin_amdgcn_wmma_f32_16x16x4_f32(
        false, av, false, bv, (short)0, acc, false, false);
    ap += 4;
    bp += 4 * 1024;
  }
  // D rows M=0 (v0, lanes 0-15) and M=1 (v1, lanes 0-15) are the batch rows
  if (half == 0) {
    atomicAdd(&h1[n],        acc[0]);
    atomicAdd(&h1[1024 + n], acc[1]);
  }
}

// ---------------------------------------------------------------------------
// fc1 bias+ReLU, fc2, fc3, softmax. Single 64-thread block.
// ---------------------------------------------------------------------------
__global__ void k_fc23(const float* __restrict__ h1, const float* __restrict__ fb1,
                       const float* __restrict__ fw2, const float* __restrict__ fb2,
                       const float* __restrict__ fw3, const float* __restrict__ fb3,
                       float* __restrict__ out) {
  __shared__ float s1[2][1024];
  __shared__ float s2[2][32];
  __shared__ float s3[2][2];
  int t = threadIdx.x;
  for (int i = t; i < 2 * 1024; i += 64) {
    int mm = i >> 10, nn = i & 1023;
    float v = h1[i] + fb1[nn];
    s1[mm][nn] = v > 0.0f ? v : 0.0f;
  }
  __syncthreads();
  {
    int mm = t >> 5, nn = t & 31;
    float acc = fb2[nn];
    for (int k = 0; k < 1024; ++k) acc += s1[mm][k] * fw2[k * 32 + nn];
    s2[mm][nn] = acc > 0.0f ? acc : 0.0f;
  }
  __syncthreads();
  if (t < 4) {
    int mm = t >> 1, nn = t & 1;
    float acc = fb3[nn];
    for (int k = 0; k < 32; ++k) acc += s2[mm][k] * fw3[k * 2 + nn];
    s3[mm][nn] = acc > 0.0f ? acc : 0.0f;
  }
  __syncthreads();
  if (t < 2) {
    float a = s3[t][0], b = s3[t][1];
    float mx = fmaxf(a, b);
    float ea = __expf(a - mx), eb = __expf(b - mx);
    float inv = 1.0f / (ea + eb);
    out[t * 2 + 0] = ea * inv;
    out[t * 2 + 1] = eb * inv;
  }
}

// ---------------------------------------------------------------------------
// Host launcher
// ---------------------------------------------------------------------------
extern "C" void kernel_launch(void* const* d_in, const int* in_sizes, int n_in,
                              void* d_out, int out_size, void* d_ws, size_t ws_size,
                              hipStream_t stream) {
  (void)in_sizes; (void)n_in; (void)out_size; (void)ws_size;
  // setup_inputs order
  const float* x   = (const float*)d_in[0];
  const int*   ROI = (const int*)  d_in[1];
  // d_in[2] = roi_layer (==3, hardcoded)
  const float* w1 = (const float*)d_in[3];  const float* b1 = (const float*)d_in[4];
  const float* g1 = (const float*)d_in[5];  const float* be1= (const float*)d_in[6];
  const float* m1 = (const float*)d_in[7];  const float* v1 = (const float*)d_in[8];
  const float* w2 = (const float*)d_in[9];  const float* b2 = (const float*)d_in[10];
  const float* g2 = (const float*)d_in[11]; const float* be2= (const float*)d_in[12];
  const float* m2 = (const float*)d_in[13]; const float* v2 = (const float*)d_in[14];
  const float* w3 = (const float*)d_in[15]; const float* b3 = (const float*)d_in[16];
  const float* g3 = (const float*)d_in[17]; const float* be3= (const float*)d_in[18];
  const float* m3 = (const float*)d_in[19]; const float* v3 = (const float*)d_in[20];
  const float* w4 = (const float*)d_in[21]; const float* b4 = (const float*)d_in[22];
  const float* g4 = (const float*)d_in[23]; const float* be4= (const float*)d_in[24];
  const float* m4 = (const float*)d_in[25]; const float* v4 = (const float*)d_in[26];
  const float* fw1= (const float*)d_in[27]; const float* fb1= (const float*)d_in[28];
  const float* fw2= (const float*)d_in[29]; const float* fb2= (const float*)d_in[30];
  const float* fw3= (const float*)d_in[31]; const float* fb3= (const float*)d_in[32];
  float* out = (float*)d_out;

  // workspace carve
  char* ws = (char*)d_ws;
  size_t off = 0;
  auto carve = [&](size_t bytes) -> void* {
    void* p = ws + off;
    off = (off + bytes + 255) & ~(size_t)255;
    return p;
  };
  float*    t1  = (float*)   carve((size_t)2*3*252*252*4);    // conv1 out (roiX)
  float*    t2  = (float*)   carve((size_t)2*20*252*252*4);   // conv2 out
  float*    h   = (float*)   carve((size_t)2*35*125*125*4);   // concat input to conv3
  float*    y3  = (float*)   carve((size_t)2*50*121*121*4);   // conv3 raw
  float*    h3  = (float*)   carve((size_t)2*50*60*60*4);     // pooled conv3
  float*    y4  = (float*)   carve((size_t)2*25*56*56*4);     // conv4 raw
  float*    h4  = (float*)   carve((size_t)2*25*28*28*4);     // pooled conv4 (flatten)
  float*    h1  = (float*)   carve((size_t)2*1024*4);         // fc1 accum
  unsigned* Wp3 = (unsigned*)carve((size_t)(896/2)*64*4);     // conv3 bf16 packed
  float*    bf3 = (float*)   carve((size_t)64*4);
  unsigned* Wp4 = (unsigned*)carve((size_t)(1280/2)*32*4);    // conv4 bf16 packed
  float*    bf4 = (float*)   carve((size_t)32*4);

  const int TB = 256;

  // conv1 + BN + ReLU -> t1 [2][3][252][252]
  k_conv5_c1<<<cdiv(2*3*252*252, TB), TB, 0, stream>>>(x, w1, b1, g1, be1, m1, v1,
                                                       t1, 2, 3, 256, 256);
  // conv2 + BN + ReLU -> t2 [2][20][252][252]
  k_conv5_c1<<<cdiv(2*20*252*252, TB), TB, 0, stream>>>(x, w2, b2, g2, be2, m2, v2,
                                                        t2, 2, 20, 256, 256);
  // pool(4,2): t2 -> h channels 0..19 [125][125]
  k_pool<<<cdiv(2*20*125*125, TB), TB, 0, stream>>>(t2, h, 2, 20, 252, 252, 4, 2,
                                                    125, 125, 35, 0, nullptr, 0);
  // ROI adaptive max pool -> h channels 20..34 (batch reversed)
  k_roipool<<<cdiv(2*15*125*125, TB), TB, 0, stream>>>(t1, ROI, h);

  // conv3 weights: K = 35*25 = 875, pad 896; N = 50, pad 64
  k_prep_w<<<cdiv((896/2)*64, TB), TB, 0, stream>>>(w3, b3, g3, be3, m3, v3,
                                                    Wp3, bf3, 50, 875, 896, 64);
  // conv4 weights: K = 50*25 = 1250, pad 1280; N = 25, pad 32
  k_prep_w<<<cdiv((1280/2)*32, TB), TB, 0, stream>>>(w4, b4, g4, be4, m4, v4,
                                                     Wp4, bf4, 25, 1250, 1280, 32);

  // conv3 WMMA: M = 121*121 = 14641, 64 M-rows per block, 4 waves (Npad=64)
  {
    dim3 grid(cdiv(121*121, 64), 2);
    k_conv_wmma<4><<<grid, 128, 0, stream>>>(h, Wp3, y3, 35, 125, 125, 121, 121,
                                             875, 896, 64, 50);
  }
  // bias + ReLU + pool(2,2): y3 -> h3 [2][50][60][60]
  k_pool<<<cdiv(2*50*60*60, TB), TB, 0, stream>>>(y3, h3, 2, 50, 121, 121, 2, 2,
                                                  60, 60, 50, 0, bf3, 1);
  // conv4 WMMA: M = 56*56 = 3136, 64 M-rows per block, 2 waves (Npad=32)
  {
    dim3 grid(cdiv(56*56, 64), 2);
    k_conv_wmma<4><<<grid, 64, 0, stream>>>(h3, Wp4, y4, 50, 60, 60, 56, 56,
                                            1250, 1280, 32, 25);
  }
  // bias + ReLU + pool(2,2): y4 -> h4 [2][25][28][28] (== flatten [2][19600])
  k_pool<<<cdiv(2*25*28*28, TB), TB, 0, stream>>>(y4, h4, 2, 25, 56, 56, 2, 2,
                                                  28, 28, 25, 0, bf4, 1);

  // fc1: zero accum, WMMA f32 16x16x4 with 8-way K split
  k_zero<<<cdiv(2048, TB), TB, 0, stream>>>(h1, 2048);
  {
    dim3 grid(64, 8);
    k_fc1_wmma<<<grid, 32, 0, stream>>>(h4, fw1, h1);
  }
  // fc1 bias/ReLU + fc2 + fc3 + softmax -> out [2][2]
  k_fc23<<<1, 64, 0, stream>>>(h1, fb1, fw2, fb2, fw3, fb3, out);
}
